// LogisticRegressionRBF_31842887532881
// MI455X (gfx1250) — matrix operand order repaired and checked by reading, not compiled, not run on hardware
//
#include <hip/hip_runtime.h>
#include <hip/hip_bf16.h>

typedef __attribute__((ext_vector_type(16))) __bf16 v16bf;
typedef __attribute__((ext_vector_type(8)))  float  v8f;

#define KOBS 65536
#define NC   4096
#define MF   64
#define BM   128            // rows of x per block (8 waves x 16 rows)
#define BN   128            // centers per LDS chunk
#define NCHUNK (NC / BN)    // 32

// ---- prep: convert centers to bf16, compute c2[n] = ||c_n||^2 --------------
__global__ void rbf_prep(const float* __restrict__ xb,
                         __bf16* __restrict__ cb,
                         float* __restrict__ c2) {
    const int row  = blockIdx.x;      // one wave32 per center row
    const int lane = threadIdx.x;     // 0..31
    const float* src = xb + (size_t)row * MF;
    float s = 0.f;
    #pragma unroll
    for (int i = lane; i < MF; i += 32) {
        float v = src[i];
        s += v * v;
        cb[(size_t)row * MF + i] = (__bf16)v;
    }
    #pragma unroll
    for (int off = 16; off > 0; off >>= 1) s += __shfl_xor(s, off, 32);
    if (lane == 0) c2[row] = s;
}

// ---- main: fused  exp(-(x2+c2-2*x@C^T)) @ w  + b  -> sigmoid ---------------
__global__ void __launch_bounds__(256) rbf_main(const float* __restrict__ x,
                                                const __bf16* __restrict__ cb,
                                                const float* __restrict__ c2,
                                                const float* __restrict__ w,
                                                const float* __restrict__ bptr,
                                                float* __restrict__ out) {
    __shared__ __align__(128) __bf16 sB[2][BN * MF];  // 2 x 16 KB double buffer
    __shared__ float sC2[2][BN];
    __shared__ float sW[2][BN];
    __shared__ float sX2[8 * 16];

    const int tid  = threadIdx.x;
    const int wave = tid >> 5;
    const int lane = tid & 31;
    const int half = lane >> 4;   // C-fragment half: rows 0..7 vs 8..15
    const int lrow = lane & 15;

    // ---- A fragments (contiguous k = 16*half + s per lane, B matches), x2 computed in f32
    const int grow = blockIdx.x * BM + wave * 16 + lrow;
    const float* xrow = x + (size_t)grow * MF;

    v16bf a0, a1;                 // K = 0..31 and K = 32..63
    float x2p = 0.f;
    #pragma unroll
    for (int jj = 0; jj < 8; ++jj) {
        const int j = jj * 2;
        // lanes 0-15: K = 0..15; lanes 16-31: K = 16..31
        const int k = j + (half << 4);
        float2 p0 = *(const float2*)(xrow + k);
        float2 p1 = *(const float2*)(xrow + k + 32);
        x2p += p0.x * p0.x + p0.y * p0.y + p1.x * p1.x + p1.y * p1.y;
        a0[j]     = (__bf16)p0.x;  a0[j + 1] = (__bf16)p0.y;
        a1[j]     = (__bf16)p1.x;  a1[j + 1] = (__bf16)p1.y;
    }
    x2p += __shfl_xor(x2p, 16, 32);            // combine the two K-halves
    if (half == 0) sX2[wave * 16 + lrow] = x2p;
    __syncthreads();

    float x2r[8];
    #pragma unroll
    for (int r = 0; r < 8; ++r) x2r[r] = sX2[wave * 16 + half * 8 + r];

    float acc[8];
    #pragma unroll
    for (int r = 0; r < 8; ++r) acc[r] = 0.f;

    // ---- async prefetch of one 16 KB bf16 center tile into LDS buffer `buf`
    // GLOBAL_LOAD_ASYNC_TO_LDS_B128: INST_OFFSET applies to BOTH the LDS and
    // the global address, so one base pair + imm offsets covers the tile.
    auto issue_tile = [&](int chunk, int buf) {
        const uint64_t ga =
            (uint64_t)((const char*)(cb + (size_t)chunk * BN * MF) + tid * 16);
        // flat-aperture rule: low 32 bits of a generic LDS pointer == LDS offset
        const uint32_t la = (uint32_t)(uintptr_t)(&sB[buf][0]) + tid * 16;
        asm volatile("global_load_async_to_lds_b128 %0, %1, off offset:0"
                     :: "v"(la), "v"(ga) : "memory");
        asm volatile("global_load_async_to_lds_b128 %0, %1, off offset:4096"
                     :: "v"(la), "v"(ga) : "memory");
        asm volatile("global_load_async_to_lds_b128 %0, %1, off offset:8192"
                     :: "v"(la), "v"(ga) : "memory");
        asm volatile("global_load_async_to_lds_b128 %0, %1, off offset:12288"
                     :: "v"(la), "v"(ga) : "memory");
        if (tid < BN) sC2[buf][tid]      = c2[chunk * BN + tid];
        else          sW[buf][tid - BN]  = w[chunk * BN + (tid - BN)];
    };

    issue_tile(0, 0);

    for (int c = 0; c < NCHUNK; ++c) {
        const int buf = c & 1;
        if (c + 1 < NCHUNK) {
            issue_tile(c + 1, buf ^ 1);
            // 4 in flight for c+1; in-order completion => chunk c is resident
            asm volatile("s_wait_asynccnt 0x4" ::: "memory");
        } else {
            asm volatile("s_wait_asynccnt 0x0" ::: "memory");
        }
        __syncthreads();   // all waves' async tiles + c2/w visible

        const __bf16* Bt  = sB[buf];
        const float*  C2t = sC2[buf];
        const float*  Wt  = sW[buf];

        #pragma unroll
        for (int sub = 0; sub < BN / 16; ++sub) {
            const int col = sub * 16 + lrow;          // lane's B column
            // B lane layout: lanes 0-15 hold K=0..15, lanes 16-31 K=16..31
            const __bf16* bp = Bt + col * MF + (half << 4);
            v16bf b0 = *(const v16bf*)(bp);           // K 0..31 piece
            v16bf b1 = *(const v16bf*)(bp + 32);      // K 32..63 piece
            v8f cfrag = {};
            cfrag = __builtin_amdgcn_wmma_f32_16x16x32_bf16(false, a0, false, b0,
                                                            (short)0, cfrag, false, false);
            cfrag = __builtin_amdgcn_wmma_f32_16x16x32_bf16(false, a1, false, b1,
                                                            (short)0, cfrag, false, false);
            const float cc2 = C2t[col];
            const float wv  = Wt[col];
            #pragma unroll
            for (int r = 0; r < 8; ++r) {
                float sq = x2r[r] + cc2 - 2.f * cfrag[r];
                acc[r] += __expf(-sq) * wv;           // fused RBF + w-reduction
            }
        }
        __syncthreads();   // tile consumed; safe to overwrite next iteration
    }

    // butterfly over the 16 lanes holding each row's columns
    #pragma unroll
    for (int r = 0; r < 8; ++r) {
        float v = acc[r];
        v += __shfl_xor(v, 1, 32);
        v += __shfl_xor(v, 2, 32);
        v += __shfl_xor(v, 4, 32);
        v += __shfl_xor(v, 8, 32);
        acc[r] = v;
    }
    if (lrow == 0) {
        const float b = bptr[0];
        const int base = blockIdx.x * BM + wave * 16 + half * 8;
        #pragma unroll
        for (int r = 0; r < 8; ++r) {
            float t = acc[r] + b;
            out[base + r] = 1.f / (1.f + __expf(-t));
        }
    }
}

extern "C" void kernel_launch(void* const* d_in, const int* in_sizes, int n_in,
                              void* d_out, int out_size, void* d_ws, size_t ws_size,
                              hipStream_t stream) {
    const float* x  = (const float*)d_in[0];   // [K, 64]
    const float* xb = (const float*)d_in[1];   // [N, 64]
    const float* w  = (const float*)d_in[2];   // [1, N]
    const float* b  = (const float*)d_in[3];   // [1]
    float* out = (float*)d_out;                // [K, 1]

    __bf16* cb = (__bf16*)d_ws;                             // N*64 bf16
    float*  c2 = (float*)((char*)d_ws + (size_t)NC * MF * sizeof(__bf16));

    rbf_prep<<<NC, 32, 0, stream>>>(xb, cb, c2);
    rbf_main<<<KOBS / BM, 256, 0, stream>>>(x, cb, c2, w, b, out);
}
